// NeuralODE_88665304859359
// MI455X (gfx1250) — matrix operand order repaired and checked by reading, not compiled, not run on hardware
//
#include <hip/hip_runtime.h>
#include <hip/hip_bf16.h>
#include <math.h>

typedef __attribute__((ext_vector_type(16))) _Float16 v16h;
typedef __attribute__((ext_vector_type(8)))  _Float16 v8h;
typedef __attribute__((ext_vector_type(8)))  float    v8f;

#define HNN 512          // hidden width (K for both WMMA layers)
#define LDH 520          // LDS row stride in halves (16B-aligned rows, bank skew)
#define MT  64           // batch rows per workgroup (4 M-subtiles of 16)

__device__ __forceinline__ float fast_tanh(float x) {
#if __has_builtin(__builtin_amdgcn_tanhf)
  return __builtin_amdgcn_tanhf(x);          // V_TANH_F32 (CDNA5 trans op)
#elif __has_builtin(__builtin_amdgcn_tanh_f32)
  return __builtin_amdgcn_tanh_f32(x);
#else
  return tanhf(x);
#endif
}

// hout[MT x (8*NT*16)] = tanh(hin[MT x 512] @ WT^T + bias)
// WT stored transposed: WT[n][k], n-major, k contiguous (K = 512).
// 8 waves; wave owns NT adjacent 16-wide N tiles; each B fragment (one L2
// read per workgroup) is reused across MS = MT/16 M-subtiles.
template<int NT>
__device__ __forceinline__ void wmma_layer(const _Float16* hin,
                                           const _Float16* __restrict__ WT,
                                           const float* __restrict__ bias,
                                           _Float16* hout,
                                           int wave, int lane) {
  constexpr int MS = MT / 16;
  const int lcol  = lane & 15;
  const int hi    = lane >> 4;     // lane half: selects K sub-offset
  const int a_off = hi * 16;       // A: K 0-15 vs 16-31
  const int b_off = hi * 16;       // B: K 0-15 vs 16-31

  const v8f zero = {};
  v8f acc[MS][NT];
#pragma unroll
  for (int ms = 0; ms < MS; ++ms)
#pragma unroll
    for (int nt = 0; nt < NT; ++nt) acc[ms][nt] = zero;

#pragma unroll 2
  for (int k0 = 0; k0 < HNN; k0 += 32) {
    // A fragments: contiguous K run {k0+a_off..+15}, matching B
    v16h a[MS];
#pragma unroll
    for (int ms = 0; ms < MS; ++ms) {
      const _Float16* arow = hin + (ms * 16 + lcol) * LDH;
      v8h alo = *(const v8h*)(arow + k0 + a_off);
      v8h ahi = *(const v8h*)(arow + k0 + a_off + 8);
#pragma unroll
      for (int i = 0; i < 8; ++i) { a[ms][i] = alo[i]; a[ms][i + 8] = ahi[i]; }
    }
#pragma unroll
    for (int nt = 0; nt < NT; ++nt) {
      const int n = (wave * NT + nt) * 16 + lcol;
      // B fragment: lane l holds column n=l%16, contiguous K run from WT row n
      v16h b = *(const v16h*)(WT + (size_t)n * HNN + k0 + b_off);
#pragma unroll
      for (int ms = 0; ms < MS; ++ms)
        acc[ms][nt] = __builtin_amdgcn_wmma_f32_16x16x32_f16(
            false, a[ms], false, b, (short)0, acc[ms][nt], false, false);
    }
  }

  // C/D layout: VGPR r -> M = 8*hi + r, N = lane&15
#pragma unroll
  for (int nt = 0; nt < NT; ++nt) {
    const int n = (wave * NT + nt) * 16 + lcol;
    const float bn = bias[n];
#pragma unroll
    for (int ms = 0; ms < MS; ++ms)
#pragma unroll
      for (int r = 0; r < 8; ++r) {
        const int m = ms * 16 + hi * 8 + r;
        hout[m * LDH + n] = (_Float16)fast_tanh(acc[ms][nt][r] + bn);
      }
  }
}

// One-time f32 -> f16 transposed weight prep into workspace.
__global__ __launch_bounds__(256) void node_prep_kernel(
    const float* __restrict__ W2, const float* __restrict__ W3,
    _Float16* __restrict__ W2T, _Float16* __restrict__ W3T) {
  const int idx = blockIdx.x * 256 + threadIdx.x;
  if (idx < 512 * 512) {                 // W2: [512 k][512 n] -> W2T[n][k]
    const int n = idx >> 9, k = idx & 511;
    W2T[idx] = (_Float16)W2[k * 512 + n];
  }
  if (idx < 256 * 512) {                 // W3: [512 k][256 n] -> W3T[n][k]
    const int n = idx >> 9, k = idx & 511;
    W3T[idx] = (_Float16)W3[k * 256 + n];
  }
}

__global__ __launch_bounds__(256) void node_rk4_kernel(
    const float* __restrict__ y0, const float* __restrict__ tp,
    const float* __restrict__ W1, const float* __restrict__ b1,
    const float* __restrict__ b2, const float* __restrict__ b3,
    const float* __restrict__ W4, const float* __restrict__ b4,
    const _Float16* __restrict__ W2T, const _Float16* __restrict__ W3T,
    float* __restrict__ out, int Bt, int T) {
  __shared__ __align__(32) _Float16 bufA[MT * LDH];   // h1 / h3   (~66.5 KB)
  __shared__ __align__(32) _Float16 bufB[MT * LDH];   // h2        (~66.5 KB)
  __shared__ float s_y[MT * 2], s_ye[MT * 2], s_acc[MT * 2], s_k[MT * 2];

  const int t    = threadIdx.x;
  const int wave = t >> 5;
  const int lane = t & 31;
  const size_t base2 = (size_t)blockIdx.x * (MT * 2);

  if (t < MT * 2) {
    const float v = y0[base2 + t];
    s_y[t]  = v;
    s_ye[t] = v;
    out[base2 + t] = v;                 // trajectory slice t=0
  }
  __syncthreads();

  for (int step = 0; step < T - 1; ++step) {
    const float dt = tp[step + 1] - tp[step];
#pragma unroll 1
    for (int e = 0; e < 4; ++e) {
      // ---- layer 1 (VALU, K=2): h1 = tanh(y @ W1 + b1) ----
#pragma unroll
      for (int cc = 0; cc < 2; ++cc) {
        const int n = t + cc * 256;
        const float w0 = W1[n], w1 = W1[HNN + n], bb = b1[n];
#pragma unroll 4
        for (int m = 0; m < MT; ++m) {
          const float v = fmaf(s_ye[m * 2 + 1], w1, fmaf(s_ye[m * 2], w0, bb));
          bufA[m * LDH + n] = (_Float16)fast_tanh(v);
        }
      }
      __syncthreads();
      // ---- layer 2 (WMMA): 512 -> 512 ----
      wmma_layer<4>(bufA, W2T, b2, bufB, wave, lane);
      __syncthreads();
      // ---- layer 3 (WMMA): 512 -> 256 ----
      wmma_layer<2>(bufB, W3T, b3, bufA, wave, lane);
      __syncthreads();
      // ---- layer 4 (VALU, N=2): k = h3 @ W4 + b4 ----
      {
        // 2 lanes per (m,j) output; cross-lane combine via shfl_xor (wave32)
        const int p = t >> 1;          // p = m*2 + j, p in [0, MT*2)
        const int q = t & 1;
        const int m = p >> 1, j = p & 1;
        float sum = 0.0f;
        const _Float16* hrow = bufA + m * LDH + q * 128;
        const float* w4c = W4 + j + (size_t)q * 256;
#pragma unroll 8
        for (int kk = 0; kk < 128; ++kk)
          sum = fmaf((float)hrow[kk], w4c[kk * 2], sum);
        sum += __shfl_xor(sum, 1, 32);
        if (q == 0) s_k[p] = sum + b4[j];
      }
      __syncthreads();
      // ---- RK4 combine ----
      if (t < MT * 2) {
        const float k = s_k[t];
        if (e == 0)      s_acc[t] = k;
        else             s_acc[t] += (e == 3 ? 1.0f : 2.0f) * k;
        if (e < 3) {
          s_ye[t] = s_y[t] + dt * (e == 2 ? 1.0f : 0.5f) * k;
        } else {
          const float yn = s_y[t] + (dt * (1.0f / 6.0f)) * s_acc[t];
          s_y[t]  = yn;
          s_ye[t] = yn;
          out[(size_t)(step + 1) * Bt * 2 + base2 + t] = yn;
        }
      }
      __syncthreads();
    }
  }
}

extern "C" void kernel_launch(void* const* d_in, const int* in_sizes, int n_in,
                              void* d_out, int out_size, void* d_ws, size_t ws_size,
                              hipStream_t stream) {
  const float* y0 = (const float*)d_in[0];
  const float* tp = (const float*)d_in[1];
  const float* W1 = (const float*)d_in[2];
  const float* b1 = (const float*)d_in[3];
  const float* W2 = (const float*)d_in[4];
  const float* b2 = (const float*)d_in[5];
  const float* W3 = (const float*)d_in[6];
  const float* b3 = (const float*)d_in[7];
  const float* W4 = (const float*)d_in[8];
  const float* b4 = (const float*)d_in[9];

  const int Bt = in_sizes[0] / 2;   // 65536
  const int T  = in_sizes[1];       // 100

  _Float16* W2T = (_Float16*)d_ws;                              // 512 KB
  _Float16* W3T = (_Float16*)((char*)d_ws + 512 * 512 * 2);     // 256 KB

  node_prep_kernel<<<1024, 256, 0, stream>>>(W2, W3, W2T, W3T);
  node_rk4_kernel<<<Bt / MT, 256, 0, stream>>>(
      y0, tp, W1, b1, b2, b3, W4, b4, W2T, W3T, (float*)d_out, Bt, T);
}
